// VimMamba_74199855005904
// MI455X (gfx1250) — compile-verified, matrix operand order checked by reading, not run on hardware
//
#include <hip/hip_runtime.h>
#include <math.h>

// ---------------- problem constants ----------------
#define Bsz   4
#define Lsz   512
#define DM    1024          // d_model
#define DS    16            // d_state
#define DCONV 4
#define DI    2048          // d_inner
#define DTR   64            // dt_rank
#define XPD   96            // dt_rank + 2*d_state
#define BL    (Bsz * Lsz)   // 2048 tokens

typedef __attribute__((ext_vector_type(16))) __bf16 v16bf;
typedef __attribute__((ext_vector_type(8)))  __bf16 bf16x8;
typedef __attribute__((ext_vector_type(8)))  float  v8f;

// ---------------- elementwise: f32 -> bf16 ----------------
__global__ __launch_bounds__(256)
void cvt_f32_bf16(const float* __restrict__ in, __bf16* __restrict__ out, int n) {
  int i = blockIdx.x * 256 + threadIdx.x;
  if (i < n) out[i] = (__bf16)in[i];
}

// Load one 16x32 bf16 fragment (ISA 16-bit A layout: lane r = row r, lanes
// r / r+16 split K {0..7,16..23} / {8..15,24..31}); works for global or LDS ptrs.
__device__ __forceinline__ v16bf load_frag(const __bf16* base) {
  bf16x8 lo = *(const bf16x8*)(base);
  bf16x8 hi = *(const bf16x8*)(base + 16);
  return __builtin_shufflevector(lo, hi, 0,1,2,3,4,5,6,7,8,9,10,11,12,13,14,15);
}

// gfx1250 async copy: 16B global -> LDS, tracked by ASYNCcnt (not LOADcnt).
// VDST = wave-relative LDS byte offset, VADDR = 64-bit global address.
__device__ __forceinline__ void async_copy_b128(unsigned lds_off, const void* gptr) {
  asm volatile("global_load_async_to_lds_b128 %0, %1, off"
               :: "v"(lds_off), "v"((unsigned long long)(size_t)gptr)
               : "memory");
}
__device__ __forceinline__ void wait_async0() {
  asm volatile("s_wait_asynccnt 0" ::: "memory");
}

// ---------------- simple WMMA GEMM (one 16x16 tile / wave) ----------------
// Used only for the narrow x_proj GEMM (N = 96).
__global__ __launch_bounds__(256)
void gemm_bf16_wmma(const __bf16* __restrict__ A, const __bf16* __restrict__ B,
                    float* __restrict__ C, int M, int N, int K,
                    int lda, int ldb, int ldc)
{
  const int lane = threadIdx.x & 31;
  const int wave = threadIdx.x >> 5;               // 0..7 -> M sub-tile
  const int m0 = (blockIdx.y * 8 + wave) * 16;
  const int n0 = blockIdx.x * 16;
  if (m0 >= M) return;

  const int r  = lane & 15;
  const int hi = (lane >> 4) << 3;                 // 0 or 8 (K sub-offset)
  const __bf16* Ap = A + (size_t)(m0 + r) * lda + hi;
  const __bf16* Bp = B + (size_t)(n0 + r) * ldb + hi;

  v8f acc = {};
  for (int k0 = 0; k0 < K; k0 += 32) {
    v16bf av = load_frag(Ap + k0);
    v16bf bv = load_frag(Bp + k0);
    acc = __builtin_amdgcn_wmma_f32_16x16x32_bf16(
        false, av, false, bv, (short)0, acc, false, false);
  }

  const int col = lane & 15;
  const int rb  = (lane >> 4) * 8;
  float* Cp = C + (size_t)(m0 + rb) * ldc + (n0 + col);
#pragma unroll
  for (int i = 0; i < 8; ++i) Cp[(size_t)i * ldc] = acc[i];
}

// ------- LDS-staged, async-double-buffered WMMA GEMM: 2x4 tiles / wave -------
// Block tile 64(M) x 256(N); 8 waves arranged 2(M) x 4(N), 32x64 per wave.
// Per 32-K step the block async-copies A(64x32)+B(256x32) bf16 (20 KB) into the
// next LDS buffer while computing 64 WMMAs from the current one; fragments are
// read with ds_load_b128.  Fence: s_wait_asynccnt 0 + workgroup barrier.
// Requires M % 64 == 0, N % 256 == 0, K % 32 == 0, 16B-aligned rows.
#define TM 64
#define TN 256
#define TK 32
__global__ __launch_bounds__(256)
void gemm_bf16_wmma_lds(const __bf16* __restrict__ A, const __bf16* __restrict__ B,
                        float* __restrict__ C, int M, int N, int K,
                        int lda, int ldb, int ldc)
{
  __shared__ __align__(16) __bf16 shA[2][TM * TK];   // 2 x 4 KB
  __shared__ __align__(16) __bf16 shB[2][TN * TK];   // 2 x 16 KB

  const int lane = threadIdx.x & 31;
  const int wave = threadIdx.x >> 5;               // 0..7
  const int wm   = wave >> 2;                      // 0..1
  const int wn   = wave & 3;                       // 0..3
  const int mblk = blockIdx.y * TM;
  const int nblk = blockIdx.x * TN;

  const char* Abase = (const char*)(A + (size_t)mblk * lda);
  const char* Bbase = (const char*)(B + (size_t)nblk * ldb);
  const size_t ldab = (size_t)lda * 2;             // row bytes
  const size_t ldbb = (size_t)ldb * 2;

  // staging assignment: 16B chunks, 4 chunks per 64B row of the LDS tile
  const int alin = wave * 32 + lane;               // 256 chunks cover A tile
  const int arow = alin >> 2;
  const int akb  = (alin & 3) * 16;                // byte offset in row

  // issue async copies for K-chunk k0 into buffer `bf`
  auto stage = [&](int bf, int k0) {
    const size_t kb0 = (size_t)k0 * 2;             // byte offset of K chunk
    async_copy_b128((unsigned)(size_t)&shA[bf][arow * TK + (akb >> 1)],
                    Abase + (size_t)arow * ldab + kb0 + akb);
#pragma unroll
    for (int q = 0; q < 4; ++q) {                  // 1024 chunks cover B tile
      int lin = wave * 128 + q * 32 + lane;
      int row = lin >> 2;
      int kb  = (lin & 3) * 16;
      async_copy_b128((unsigned)(size_t)&shB[bf][row * TK + (kb >> 1)],
                      Bbase + (size_t)row * ldbb + kb0 + kb);
    }
  };

  const int r  = lane & 15;
  const int hi = (lane >> 4) << 3;                 // 0 or 8

  v8f acc[2][4];
#pragma unroll
  for (int i = 0; i < 2; ++i)
#pragma unroll
    for (int j = 0; j < 4; ++j) acc[i][j] = (v8f){};

  stage(0, 0);
  wait_async0();
  __syncthreads();

  int buf = 0;
  for (int k0 = 0; k0 < K; k0 += TK) {
    if (k0 + TK < K) stage(buf ^ 1, k0 + TK);      // overlap copy with compute

    v16bf a[2], b[4];
#pragma unroll
    for (int i = 0; i < 2; ++i)
      a[i] = load_frag(&shA[buf][(wm * 32 + i * 16 + r) * TK + hi]);
#pragma unroll
    for (int j = 0; j < 4; ++j)
      b[j] = load_frag(&shB[buf][(wn * 64 + j * 16 + r) * TK + hi]);

#pragma unroll
    for (int i = 0; i < 2; ++i)
#pragma unroll
      for (int j = 0; j < 4; ++j)
        acc[i][j] = __builtin_amdgcn_wmma_f32_16x16x32_bf16(
            false, a[i], false, b[j], (short)0, acc[i][j], false, false);

    wait_async0();                                  // own async copies landed
    __syncthreads();                                // whole block's copies visible
    buf ^= 1;
  }

  const int col = lane & 15;
  const int rb  = (lane >> 4) * 8;
#pragma unroll
  for (int i = 0; i < 2; ++i) {
#pragma unroll
    for (int j = 0; j < 4; ++j) {
      float* Cp = C + (size_t)(mblk + wm * 32 + i * 16 + rb) * ldc
                    + (nblk + wn * 64 + j * 16 + col);
#pragma unroll
      for (int q = 0; q < 8; ++q) Cp[(size_t)q * ldc] = acc[i][j][q];
    }
  }
}

// ---------------- depthwise causal conv (k=4) + bias, dual f32/bf16 out ----------------
__global__ __launch_bounds__(256)
void conv_kernel(const float* __restrict__ xz, const float* __restrict__ w,
                 const float* __restrict__ bias, float* __restrict__ xc,
                 __bf16* __restrict__ xc_bf, int dir)
{
  int t = blockIdx.x * 256 + threadIdx.x;
  if (t >= BL * DI) return;
  int d  = t % DI;
  int bl = t / DI;
  int b  = bl / Lsz;
  int l  = bl % Lsz;
  float acc = bias[d];
#pragma unroll
  for (int k = 0; k < DCONV; ++k) {
    int j = l - (DCONV - 1) + k;
    if (j >= 0) {
      int phys = dir ? (Lsz - 1 - j) : j;
      acc = fmaf(w[d * DCONV + k], xz[((size_t)(b * Lsz + phys)) * (2 * DI) + d], acc);
    }
  }
  xc[t]    = acc;
  xc_bf[t] = (__bf16)acc;
}

// ---------------- dt epilogue: clip + double-bias softplus (in place) ----------------
__global__ __launch_bounds__(256)
void delta_kernel(float* __restrict__ dt, const float* __restrict__ bias)
{
  int t = blockIdx.x * 256 + threadIdx.x;
  if (t >= BL * DI) return;
  int d  = t % DI;
  float bv = bias[d];
  float v  = dt[t] + bv;                 // dt = x_dbl@W^T + b
  v = fminf(1.0f, fmaxf(1e-5f, v));      // clip
  float x = v + bv;                      // softplus(dt + b)  (bias added twice, per ref)
  dt[t] = (x > 20.0f) ? x : log1pf(__expf(x));
}

// ---------------- selective scan: one thread per (b, d) channel ----------------
__global__ __launch_bounds__(256)
void scan_kernel(const float* __restrict__ delta, const float* __restrict__ xc,
                 const float* __restrict__ xdbl, const float* __restrict__ A_log,
                 const float* __restrict__ Dv, float* __restrict__ y,
                 int dir, int accumulate)
{
  int t = blockIdx.x * 256 + threadIdx.x;
  if (t >= Bsz * DI) return;
  int b = t / DI;
  int d = t % DI;

  float An[DS], h[DS];
#pragma unroll
  for (int n = 0; n < DS; ++n) { An[n] = -__expf(A_log[d * DS + n]); h[n] = 0.0f; }
  float Dd = Dv[d];

  for (int lp = 0; lp < Lsz; ++lp) {
    size_t idx = ((size_t)(b * Lsz + lp)) * DI + d;
    float dl = delta[idx];
    float u  = fminf(10.0f, fmaxf(-10.0f, xc[idx]));
    const float* xd = xdbl + (size_t)(b * Lsz + lp) * XPD;
    float du = dl * u;
    float yv = 0.0f;
#pragma unroll
    for (int n = 0; n < DS; ++n) {
      float dA = __expf(dl * An[n]);
      h[n] = fmaf(dA, h[n], du * xd[DTR + n]);        // B_t
      yv   = fmaf(h[n], xd[DTR + DS + n], yv);        // C_t
    }
    yv = fmaf(u, Dd, yv);
    int phys = dir ? (Lsz - 1 - lp) : lp;             // un-reverse bwd output
    size_t oidx = ((size_t)(b * Lsz + phys)) * DI + d;
    if (accumulate) y[oidx] += yv; else y[oidx] = yv;
  }
}

// ---------------- gating: y = (y_fwd + y_bwd) * silu(z), bf16 out ----------------
__global__ __launch_bounds__(256)
void gate_kernel(const float* __restrict__ y, const float* __restrict__ xz,
                 __bf16* __restrict__ y_bf)
{
  int t = blockIdx.x * 256 + threadIdx.x;
  if (t >= BL * DI) return;
  int d  = t % DI;
  int bl = t / DI;
  float z = xz[(size_t)bl * (2 * DI) + DI + d];
  float s = z / (1.0f + __expf(-z));
  y_bf[t] = (__bf16)(y[t] * s);
}

// ---------------- nan_to_num(nan=0, posinf=1, neginf=-1), in place ----------------
__global__ __launch_bounds__(256)
void nan_kernel(float* __restrict__ out, int n)
{
  int i = blockIdx.x * 256 + threadIdx.x;
  if (i >= n) return;
  float v = out[i];
  if (isnan(v)) v = 0.0f;
  else if (isinf(v)) v = (v > 0.0f) ? 1.0f : -1.0f;
  out[i] = v;
}

// ---------------- host orchestration ----------------
extern "C" void kernel_launch(void* const* d_in, const int* in_sizes, int n_in,
                              void* d_out, int out_size, void* d_ws, size_t ws_size,
                              hipStream_t stream) {
  const float* hs        = (const float*)d_in[0];
  const float* win       = (const float*)d_in[1];
  const float* convw[2]  = {(const float*)d_in[2],  (const float*)d_in[9]};
  const float* convb[2]  = {(const float*)d_in[3],  (const float*)d_in[10]};
  const float* xprojw[2] = {(const float*)d_in[4],  (const float*)d_in[11]};
  const float* dtw[2]    = {(const float*)d_in[5],  (const float*)d_in[12]};
  const float* dtb[2]    = {(const float*)d_in[6],  (const float*)d_in[13]};
  const float* alog[2]   = {(const float*)d_in[7],  (const float*)d_in[14]};
  const float* dvec[2]   = {(const float*)d_in[8],  (const float*)d_in[15]};
  const float* wout      = (const float*)d_in[16];
  float* out = (float*)d_out;

  // bump allocator over d_ws
  char* p = (char*)d_ws;
  auto alloc = [&](size_t bytes) -> char* {
    char* r = p; p += (bytes + 255) & ~(size_t)255; return r;
  };

  float*  xz      = (float*) alloc((size_t)BL * 2 * DI * sizeof(float));
  __bf16* hs_bf   = (__bf16*)alloc((size_t)BL * DM * 2);
  __bf16* wi_bf   = (__bf16*)alloc((size_t)2 * DI * DM * 2);
  __bf16* wo_bf   = (__bf16*)alloc((size_t)DM * DI * 2);
  __bf16* wx_bf   = (__bf16*)alloc((size_t)2 * XPD * DI * 2);
  __bf16* wd_bf   = (__bf16*)alloc((size_t)2 * DI * DTR * 2);
  float*  xc      = (float*) alloc((size_t)BL * DI * sizeof(float));
  __bf16* xc_bf   = (__bf16*)alloc((size_t)BL * DI * 2);
  float*  xdbl    = (float*) alloc((size_t)BL * XPD * sizeof(float));
  __bf16* xdbl_bf = (__bf16*)alloc((size_t)BL * XPD * 2);
  float*  dt      = (float*) alloc((size_t)BL * DI * sizeof(float));
  float*  ybuf    = (float*) alloc((size_t)BL * DI * sizeof(float));
  __bf16* y_bf    = (__bf16*)alloc((size_t)BL * DI * 2);

  auto cvt = [&](const float* src, __bf16* dst, int n) {
    cvt_f32_bf16<<<dim3((n + 255) / 256), 256, 0, stream>>>(src, dst, n);
  };
  // LDS/async path: M % 64 == 0, N % 256 == 0
  auto gemm_big = [&](const __bf16* A, const __bf16* B, float* C,
                      int M, int N, int K, int lda, int ldb, int ldc) {
    dim3 grid(N / 256, M / 64);
    gemm_bf16_wmma_lds<<<grid, 256, 0, stream>>>(A, B, C, M, N, K, lda, ldb, ldc);
  };
  // fallback path for narrow N
  auto gemm_small = [&](const __bf16* A, const __bf16* B, float* C,
                        int M, int N, int K, int lda, int ldb, int ldc) {
    dim3 grid(N / 16, (M + 127) / 128);
    gemm_bf16_wmma<<<grid, 256, 0, stream>>>(A, B, C, M, N, K, lda, ldb, ldc);
  };

  const int NE = BL * DI;                  // 4M elementwise threads
  dim3 gew((NE + 255) / 256);

  // ---- precision conversion of activations + all weights ----
  cvt(hs,   hs_bf, BL * DM);
  cvt(win,  wi_bf, 2 * DI * DM);
  cvt(wout, wo_bf, DM * DI);
  cvt(xprojw[0], wx_bf,            XPD * DI);
  cvt(xprojw[1], wx_bf + XPD * DI, XPD * DI);
  cvt(dtw[0],    wd_bf,            DI * DTR);
  cvt(dtw[1],    wd_bf + DI * DTR, DI * DTR);

  // ---- in_proj: xz[BL][4096] = hs[BL][1024] @ win[4096][1024]^T ----
  gemm_big(hs_bf, wi_bf, xz, BL, 2 * DI, DM, DM, DM, 2 * DI);

  // ---- two directions (sequential, shared scratch) ----
  for (int dir = 0; dir < 2; ++dir) {
    conv_kernel<<<gew, 256, 0, stream>>>(xz, convw[dir], convb[dir], xc, xc_bf, dir);

    // x_dbl[BL][96] = xc[BL][2048] @ xproj[96][2048]^T   (narrow N -> simple kernel)
    gemm_small(xc_bf, wx_bf + (size_t)dir * XPD * DI, xdbl, BL, XPD, DI, DI, DI, XPD);
    cvt(xdbl, xdbl_bf, BL * XPD);

    // dt[BL][2048] = x_dbl[:, :64] @ dtw[2048][64]^T   (lda = 96 row stride)
    gemm_big(xdbl_bf, wd_bf + (size_t)dir * DI * DTR, dt, BL, DI, DTR, XPD, DTR, DI);
    delta_kernel<<<gew, 256, 0, stream>>>(dt, dtb[dir]);

    scan_kernel<<<dim3((Bsz * DI + 255) / 256), 256, 0, stream>>>(
        dt, xc, xdbl, alog[dir], dvec[dir], ybuf, dir, /*accumulate=*/dir);
  }

  // ---- gate + out_proj + nan_to_num ----
  gate_kernel<<<gew, 256, 0, stream>>>(ybuf, xz, y_bf);
  gemm_big(y_bf, wo_bf, out, BL, DM, DI, DI, DI, DM);
  nan_kernel<<<dim3((out_size + 255) / 256), 256, 0, stream>>>(out, out_size);
}
